// HLiuAttn_72748156059848
// MI455X (gfx1250) — matrix operand order repaired and checked by reading, not compiled, not run on hardware
//
#include <hip/hip_runtime.h>

// ---------------------------------------------------------------------------
// CDNA5 (gfx1250) fused QKV + attention.  B=4, N=2048, D=1024, fp32 in/out.
// GEMMs run on v_wmma_f32_16x16x32_bf16 (wave32).
// ---------------------------------------------------------------------------

typedef __attribute__((ext_vector_type(16))) __bf16 v16bf;
typedef __attribute__((ext_vector_type(8)))  float  v8f;

union VF8   { v8f v; float f[8]; };
union Frag  { v16bf v; uint4 q[2]; };

static __device__ __forceinline__ v8f wmma_bf16(v16bf a, v16bf b, v8f c) {
  // (neg_a, A, neg_b, B, c_mod, C, reuse_a, reuse_b)
  return __builtin_amdgcn_wmma_f32_16x16x32_bf16(false, a, false, b,
                                                 (short)0, c, false, false);
}

static __device__ __forceinline__ unsigned bfbits(float x) {
  union { float f; unsigned u; } v; v.f = x; return v.u;
}

// f32 -> bf16 (round-half-up bias; v_perm_b32 splices two high halves).
// (probe: __builtin_amdgcn_cvt_pk_bf16_f32 not available on this toolchain)
static __device__ __forceinline__ unsigned pk2(float a, float b) {
  return __builtin_amdgcn_perm(bfbits(b) + 0x8000u, bfbits(a) + 0x8000u,
                               0x07060302u);
}
static __device__ __forceinline__ unsigned short f2bf(float x) {
  return (unsigned short)((bfbits(x) + 0x8000u) >> 16);
}

static __device__ __forceinline__ uint4 pack8(float4 a, float4 b) {
  uint4 r;
  r.x = pk2(a.x, a.y); r.y = pk2(a.z, a.w);
  r.z = pk2(b.x, b.y); r.w = pk2(b.z, b.w);
  return r;
}

// A-matrix fragment: 16x32 bf16 tile, row-major source with stride ld (elems).
// ISA layout: lanes 0-15 -> M=lane, K elems {kb..kb+7, kb+16..kb+23}, kb=(lane>>4)*8
static __device__ __forceinline__ v16bf load_a_frag(const unsigned short* base, int ld) {
  const int lane = threadIdx.x & 31;
  const int row  = lane & 15;
  const int kb   = (lane >> 4) * 8;
  Frag f;
  f.q[0] = *reinterpret_cast<const uint4*>(base + (size_t)row * ld + kb);
  f.q[1] = *reinterpret_cast<const uint4*>(base + (size_t)row * ld + kb + 16);
  return f.v;
}

// B-matrix fragment: 32x16 bf16, source stored as B^T [n][k] row-major, stride ld.
// ISA layout: lanes 0-15 -> N=lane, K elems {kb..kb+7, kb+16..kb+23}, kb=(lane>>4)*8
static __device__ __forceinline__ v16bf load_b_frag(const unsigned short* base, int ld) {
  const int lane = threadIdx.x & 31;
  const int col  = lane & 15;
  const int kb   = (lane >> 4) * 8;
  Frag f;
  f.q[0] = *reinterpret_cast<const uint4*>(base + (size_t)col * ld + kb);
  f.q[1] = *reinterpret_cast<const uint4*>(base + (size_t)col * ld + kb + 16);
  return f.v;
}

static __device__ __forceinline__ v8f zero8() {
  v8f z = {0.f, 0.f, 0.f, 0.f, 0.f, 0.f, 0.f, 0.f};
  return z;
}

// ---------------------------------------------------------------------------
// Kernel 1: qkv = x @ w^T  (fp32 -> bf16).  Q,K row-major [8192][1024];
// V stored transposed per batch: Vt[b][d][n]  ([4][1024][2048]).
// Grid (64, 24), block 256.  WG tile 128(M) x 128(E), K-step 32.
// 8 waves as 4(m) x 2(e); wave tile 32x64 -> 8 WMMAs per 6 frag-loads.
// ---------------------------------------------------------------------------
__global__ __launch_bounds__(256)
void HLiuAttn_qkv_kernel(const float* __restrict__ x,
                         const float* __restrict__ w,
                         unsigned short* __restrict__ Qb,
                         unsigned short* __restrict__ Kb,
                         unsigned short* __restrict__ Vt) {
  __shared__ __align__(16) unsigned short xs[128 * 32];   // x tile, bf16 (8 KB)
  __shared__ __align__(16) unsigned short wbs[128 * 32];  // w tile, bf16 (8 KB)

  const int t    = threadIdx.x;
  const int lane = t & 31;
  const int wid  = t >> 5;
  const int wm   = wid & 3;      // 4 m-groups of 32 rows
  const int we   = wid >> 2;     // 2 e-groups of 64 cols
  const int mWG  = blockIdx.x * 128;
  const int eWG  = blockIdx.y * 128;

  const int sr = t >> 1, sh = (t & 1) * 16;   // staging: 16 f32 per thread

  VF8 acc[2][4];
#pragma unroll
  for (int i = 0; i < 2; ++i)
#pragma unroll
    for (int j = 0; j < 4; ++j) acc[i][j].v = zero8();

  for (int k0 = 0; k0 < 1024; k0 += 32) {
    // stage x[128][32] -> bf16 LDS
    {
      const float4* src = reinterpret_cast<const float4*>(
          x + (size_t)(mWG + sr) * 1024 + k0 + sh);
      float4 f0 = src[0], f1 = src[1], f2 = src[2], f3 = src[3];
      uint4* dst = reinterpret_cast<uint4*>(&xs[sr * 32 + sh]);
      dst[0] = pack8(f0, f1);
      dst[1] = pack8(f2, f3);
    }
    // stage w[128][32] -> bf16 LDS (already the B^T [e][k] layout we need)
    {
      const float4* src = reinterpret_cast<const float4*>(
          w + (size_t)(eWG + sr) * 1024 + k0 + sh);
      float4 g0 = src[0], g1 = src[1], g2 = src[2], g3 = src[3];
      uint4* dst = reinterpret_cast<uint4*>(&wbs[sr * 32 + sh]);
      dst[0] = pack8(g0, g1);
      dst[1] = pack8(g2, g3);
    }
    __syncthreads();

    v16bf a0 = load_a_frag(xs + (size_t)(wm * 32 + 0)  * 32, 32);
    v16bf a1 = load_a_frag(xs + (size_t)(wm * 32 + 16) * 32, 32);
#pragma unroll
    for (int et = 0; et < 4; ++et) {
      v16bf bb = load_b_frag(wbs + (size_t)(we * 64 + et * 16) * 32, 32);
      acc[0][et].v = wmma_bf16(a0, bb, acc[0][et].v);
      acc[1][et].v = wmma_bf16(a1, bb, acc[1][et].v);
    }
    __syncthreads();
  }

  // Store.  C/D layout: col = lane&15, rows (lane>>4)*8 + r.
  const int col  = lane & 15;
  const int rr0  = (lane >> 4) * 8;
  const int sect = eWG >> 10;              // 0=Q, 1=K, 2=V (tile never straddles)
  const int eloc = (eWG & 1023) + we * 64;

#pragma unroll
  for (int mt = 0; mt < 2; ++mt) {
#pragma unroll
    for (int et = 0; et < 4; ++et) {
      const int m = mWG + wm * 32 + mt * 16 + rr0;
      const int e = eloc + et * 16 + col;
      if (sect == 0) {
#pragma unroll
        for (int r = 0; r < 8; ++r)
          Qb[(size_t)(m + r) * 1024 + e] = f2bf(acc[mt][et].f[r]);
      } else if (sect == 1) {
#pragma unroll
        for (int r = 0; r < 8; ++r)
          Kb[(size_t)(m + r) * 1024 + e] = f2bf(acc[mt][et].f[r]);
      } else {
        // transposed V store: 8 consecutive n -> one packed 16B store
        const int bb = m >> 11;            // batch
        const int n  = m & 2047;
        uint4 pk;
        pk.x = pk2(acc[mt][et].f[0], acc[mt][et].f[1]);
        pk.y = pk2(acc[mt][et].f[2], acc[mt][et].f[3]);
        pk.z = pk2(acc[mt][et].f[4], acc[mt][et].f[5]);
        pk.w = pk2(acc[mt][et].f[6], acc[mt][et].f[7]);
        *reinterpret_cast<uint4*>(&Vt[((size_t)bb * 1024 + e) * 2048 + n]) = pk;
      }
    }
  }
}

// ---------------------------------------------------------------------------
// Kernel 2: flash attention over full D=1024, 32 query rows per WG.
// Grid 256 (= 4 batches * 64 query tiles of 32), block 256 (8 waves).
// Wave w owns D-slice [w*128, w*128+128): keeps two 16-row Q slices in
// A-frags, accumulates 2 x (16x128) of the output.  Every K/V B-fragment is
// loaded once and feeds two WMMAs.  S reduced across waves via ds_add_f32.
// ---------------------------------------------------------------------------
__global__ __launch_bounds__(256)
void HLiuAttn_attn_kernel(const unsigned short* __restrict__ Qb,
                          const unsigned short* __restrict__ Kb,
                          const unsigned short* __restrict__ Vt,
                          float* __restrict__ out) {
  constexpr int   N     = 2048;
  constexpr int   D     = 1024;
  constexpr float scale = 0.03125f;   // 1024^-0.5

  __shared__ __align__(16) float          Sbuf[32 * 64];
  __shared__ __align__(16) unsigned short Pbuf[32 * 64];
  __shared__ __align__(16) float          alphaB[32];
  __shared__ __align__(16) float          lB[32];

  const int t    = threadIdx.x;
  const int lane = t & 31;
  const int wid  = t >> 5;
  const int b    = blockIdx.x >> 6;
  const int q0   = (blockIdx.x & 63) << 5;
  const int col  = lane & 15;
  const int r0   = (lane >> 4) * 8;
  const int dW   = wid * 128;

  // resident Q slices: 2 x (16 rows x 128 d), as 4 A-fragments each
  const unsigned short* Qp = Qb + ((size_t)b * N + q0) * D + dW;
  v16bf qf[2][4];
#pragma unroll
  for (int qt = 0; qt < 2; ++qt)
#pragma unroll
    for (int j = 0; j < 4; ++j)
      qf[qt][j] = load_a_frag(Qp + (size_t)qt * 16 * D + j * 32, D);

  VF8 acc[2][8];
#pragma unroll
  for (int qt = 0; qt < 2; ++qt)
#pragma unroll
    for (int i = 0; i < 8; ++i) acc[qt][i].v = zero8();

  float mst[4] = {-1e30f, -1e30f, -1e30f, -1e30f};
  float lst[4] = {0.f, 0.f, 0.f, 0.f};

  for (int kb0 = 0; kb0 < N; kb0 += 64) {
    // ---- zero the shared S tile (32x64 f32) ----
#pragma unroll
    for (int i = 0; i < 8; ++i) Sbuf[t + 256 * i] = 0.f;
    __syncthreads();

    // ---- partial S = Q_slices @ K_slice^T; each B-frag feeds 2 WMMAs ----
    const unsigned short* Kp = Kb + ((size_t)b * N + kb0) * D + dW;
    VF8 s[2][4];
#pragma unroll
    for (int qt = 0; qt < 2; ++qt)
#pragma unroll
      for (int nt = 0; nt < 4; ++nt) s[qt][nt].v = zero8();
#pragma unroll
    for (int nt = 0; nt < 4; ++nt) {
#pragma unroll
      for (int j = 0; j < 4; ++j) {
        v16bf bf = load_b_frag(Kp + (size_t)(nt * 16) * D + j * 32, D);
        s[0][nt].v = wmma_bf16(qf[0][j], bf, s[0][nt].v);
        s[1][nt].v = wmma_bf16(qf[1][j], bf, s[1][nt].v);
      }
    }
    if (kb0 + 64 < N) {
      __builtin_prefetch(Kp + (size_t)64 * D, 0, 0);
      __builtin_prefetch(Vt + ((size_t)b * D + dW) * N + kb0 + 64, 0, 0);
    }
    // cross-wave reduction of S into LDS
#pragma unroll
    for (int qt = 0; qt < 2; ++qt)
#pragma unroll
      for (int nt = 0; nt < 4; ++nt)
#pragma unroll
        for (int r = 0; r < 8; ++r)
          atomicAdd(&Sbuf[(qt * 16 + r0 + r) * 64 + nt * 16 + col],
                    s[qt][nt].f[r]);
    __syncthreads();

    // ---- online softmax: wave w owns rows 4w .. 4w+3 (of 32) ----
#pragma unroll
    for (int i = 0; i < 4; ++i) {
      const int r = wid * 4 + i;
      float s0 = Sbuf[r * 64 + lane] * scale;
      float s1 = Sbuf[r * 64 + 32 + lane] * scale;
      float mx = fmaxf(s0, s1);
#pragma unroll
      for (int o = 16; o > 0; o >>= 1) mx = fmaxf(mx, __shfl_xor(mx, o, 32));
      const float mnew = fmaxf(mst[i], mx);
      const float p0 = __expf(s0 - mnew);
      const float p1 = __expf(s1 - mnew);
      Pbuf[r * 64 + lane]      = f2bf(p0);
      Pbuf[r * 64 + 32 + lane] = f2bf(p1);
      float sum = p0 + p1;
#pragma unroll
      for (int o = 16; o > 0; o >>= 1) sum += __shfl_xor(sum, o, 32);
      const float alpha = __expf(mst[i] - mnew);
      lst[i] = lst[i] * alpha + sum;
      mst[i] = mnew;
      if (lane == 0) { alphaB[r] = alpha; lB[r] = lst[i]; }
    }
    __syncthreads();

    // ---- rescale accumulators, then acc += P @ V_slice ----
#pragma unroll
    for (int qt = 0; qt < 2; ++qt) {
      const float4 a0 = *reinterpret_cast<const float4*>(&alphaB[qt * 16 + r0]);
      const float4 a1 = *reinterpret_cast<const float4*>(&alphaB[qt * 16 + r0 + 4]);
      const float alv[8] = {a0.x, a0.y, a0.z, a0.w, a1.x, a1.y, a1.z, a1.w};
#pragma unroll
      for (int n8 = 0; n8 < 8; ++n8)
#pragma unroll
        for (int r = 0; r < 8; ++r) acc[qt][n8].f[r] *= alv[r];
    }

    const unsigned short* Vp = Vt + ((size_t)b * D + dW) * N + kb0;
#pragma unroll
    for (int kc = 0; kc < 2; ++kc) {
      v16bf pa0 = load_a_frag(Pbuf + 0 * 16 * 64 + kc * 32, 64);
      v16bf pa1 = load_a_frag(Pbuf + 1 * 16 * 64 + kc * 32, 64);
#pragma unroll
      for (int n8 = 0; n8 < 8; ++n8) {
        v16bf vb = load_b_frag(Vp + (size_t)(n8 * 16) * N + kc * 32, N);
        acc[0][n8].v = wmma_bf16(pa0, vb, acc[0][n8].v);
        acc[1][n8].v = wmma_bf16(pa1, vb, acc[1][n8].v);
      }
    }
    __syncthreads();
  }

  // ---- normalize and store fp32 output ----
#pragma unroll
  for (int qt = 0; qt < 2; ++qt) {
    const float4 l0 = *reinterpret_cast<const float4*>(&lB[qt * 16 + r0]);
    const float4 l1 = *reinterpret_cast<const float4*>(&lB[qt * 16 + r0 + 4]);
    const float inv[8] = {1.f / l0.x, 1.f / l0.y, 1.f / l0.z, 1.f / l0.w,
                          1.f / l1.x, 1.f / l1.y, 1.f / l1.z, 1.f / l1.w};
#pragma unroll
    for (int n8 = 0; n8 < 8; ++n8) {
      const int d0 = dW + n8 * 16 + col;
#pragma unroll
      for (int r = 0; r < 8; ++r)
        out[((size_t)b * N + q0 + qt * 16 + r0 + r) * D + d0] =
            acc[qt][n8].f[r] * inv[r];
    }
  }
}

// ---------------------------------------------------------------------------
extern "C" void kernel_launch(void* const* d_in, const int* in_sizes, int n_in,
                              void* d_out, int out_size, void* d_ws, size_t ws_size,
                              hipStream_t stream) {
  (void)in_sizes; (void)n_in; (void)out_size; (void)ws_size;
  const float* x = (const float*)d_in[0];      // [4,2048,1024]
  const float* w = (const float*)d_in[1];      // [3072,1024]
  float* out = (float*)d_out;                  // [4,2048,1024]

  unsigned short* Qb = (unsigned short*)d_ws;                  // 16 MB
  unsigned short* Kb = Qb + (size_t)8192 * 1024;               // 16 MB
  unsigned short* Vt = Kb + (size_t)8192 * 1024;               // 16 MB ([4][1024][2048])

  HLiuAttn_qkv_kernel<<<dim3(64, 24), 256, 0, stream>>>(x, w, Qb, Kb, Vt);
  HLiuAttn_attn_kernel<<<dim3(256), 256, 0, stream>>>(Qb, Kb, Vt, out);
}
